// Transformer_1632087572872
// MI455X (gfx1250) — compile-verified, matrix operand order checked by reading, not run on hardware
//
#include <hip/hip_runtime.h>
#include <stdint.h>

// ---------------------------------------------------------------------------
// CDNA5 (gfx1250) transformer forward: bf16 WMMA GEMMs + flash attention.
// Explicit address spaces (global_load_b128 / ds_store_b128, never flat_*),
// fragment-packed weights, LDS-double-buffered A, register-double-buffered B.
// ---------------------------------------------------------------------------

typedef __attribute__((ext_vector_type(16))) __bf16       v16bf;
typedef __attribute__((ext_vector_type(8)))  float        v8f;
typedef __attribute__((ext_vector_type(4)))  unsigned int v4u;   // 16B chunk

// Address-space-qualified pointer types (1 = global, 3 = LDS).
typedef const __bf16 __attribute__((address_space(1))) *gcbf_t;
typedef       __bf16 __attribute__((address_space(1))) *gbf_t;
typedef const v4u    __attribute__((address_space(1))) *gcu4_t;
typedef       v4u    __attribute__((address_space(1))) *gu4_t;
typedef const v16bf  __attribute__((address_space(1))) *gcv16_t;
typedef const float  __attribute__((address_space(1))) *gcf_t;
typedef       float  __attribute__((address_space(1))) *gf_t;
typedef const unsigned char __attribute__((address_space(1))) *gcu8_t;
typedef const __bf16 __attribute__((address_space(3))) *lcbf_t;
typedef       __bf16 __attribute__((address_space(3))) *lbf_t;
typedef       v4u    __attribute__((address_space(3))) *lu4_t;
typedef const v16bf  __attribute__((address_space(3))) *lcv16_t;

#define NTOK   4096
#define DIMX   768
#define QKVW   2304
#define FFW    3072
#define NHEAD  12
#define DHEAD  64
#define SEQ    2048

__device__ __forceinline__ v8f wmma_bf16(v16bf a, v16bf b, v8f c) {
  // v_wmma_f32_16x16x32_bf16  D = A(16x32) * B(32x16) + C(16x16 f32)
  return __builtin_amdgcn_wmma_f32_16x16x32_bf16(false, a, false, b,
                                                 (short)0, c, false, false);
}

// A-matrix 16x32 bf16 fragment from a row-major LDS tile.
// Lane L: M = L%16; element e -> K = (e/8)*16 + (L/16)*8 + e%8.
// The 16 elements form two contiguous 16B runs -> 2x ds_load_b128.
__device__ __forceinline__ v16bf frag_row(lcbf_t p, int ld) {
  int L = threadIdx.x & 31;
  int m = L & 15, g = L >> 4;
  v16bf f;
#pragma unroll
  for (int e = 0; e < 16; ++e) {
    int k = ((e >> 3) << 4) + (g << 3) + (e & 7);
    f[e] = p[m * ld + k];
  }
  return f;
}

// ---------------------------------------------------------------------------
// Weight repack: W f32 [K,N] -> bf16 WMMA B-fragment layout
//   Wp[nb][kb][lane][e] ; nb = n/16, kb = k/32,
//   lane = (n%16) + 16*g, k%32 = (e/8)*16 + g*8 + e%8.
// ---------------------------------------------------------------------------
__global__ __launch_bounds__(64) void repack_w_kernel(
    const float* __restrict__ W, __bf16* __restrict__ Wp, int K, int N) {
  gcf_t Wg = (gcf_t)W;
  int nb = blockIdx.x, kb = blockIdx.y;
  int t = threadIdx.x;
  int L = t & 31, half = t >> 5;
  int g = L >> 4;
  int n = nb * 16 + (L & 15);
  int kbase = kb * 32 + half * 16 + g * 8;
  union { __bf16 h[8]; v4u v; } u;
#pragma unroll
  for (int i = 0; i < 8; ++i)
    u.h[i] = (__bf16)Wg[(size_t)(kbase + i) * N + n];
  size_t o = (((size_t)nb * (K >> 5) + kb) * 32 + L) * 16 + half * 8;
  *(gu4_t)((gbf_t)Wp + o) = u.v;
}

// ---------------------------------------------------------------------------
// LayerNorm: one block (256 threads) per row of 768.
// mode 0: write bf16 (WMMA feed). mode 1: write f32 (final output).
// ---------------------------------------------------------------------------
__global__ __launch_bounds__(256) void ln_kernel(
    const float* __restrict__ x, const float* __restrict__ gam,
    const float* __restrict__ bet, __bf16* __restrict__ outb,
    float* __restrict__ outf, int mode) {
  __shared__ float red[256];
  int row = blockIdx.x;
  int tid = threadIdx.x;
  gcf_t xr   = (gcf_t)x + (size_t)row * DIMX;
  gcf_t gamg = (gcf_t)gam;
  gcf_t betg = (gcf_t)bet;

  float s = 0.f;
  for (int c = tid; c < DIMX; c += 256) s += xr[c];
  red[tid] = s;
  __syncthreads();
  for (int off = 128; off > 0; off >>= 1) {
    if (tid < off) red[tid] += red[tid + off];
    __syncthreads();
  }
  float mean = red[0] * (1.0f / DIMX);
  __syncthreads();

  float v = 0.f;
  for (int c = tid; c < DIMX; c += 256) {
    float d = xr[c] - mean;
    v += d * d;
  }
  red[tid] = v;
  __syncthreads();
  for (int off = 128; off > 0; off >>= 1) {
    if (tid < off) red[tid] += red[tid + off];
    __syncthreads();
  }
  float rstd = rsqrtf(red[0] * (1.0f / DIMX) + 1e-5f);

  for (int c = tid; c < DIMX; c += 256) {
    float y = (xr[c] - mean) * rstd * gamg[c] + betg[c];
    if (mode == 0) ((gbf_t)outb)[(size_t)row * DIMX + c] = (__bf16)y;
    else           ((gf_t)outf)[(size_t)row * DIMX + c] = y;
  }
}

// ---------------------------------------------------------------------------
// GEMM: C[M,N] = A(bf16 row-major)[M,K] @ Bp(bf16 fragment-packed)[K,N].
// Block: 256 threads = 8 waves; tile 128x128x32; wave subtile 32x64 (2x4 wmma).
// A double-buffered in LDS; B fragments register-double-buffered (loaded one
// K-step ahead, K-loop unrolled by 2 so the buffers alternate by code
// position). KB (=K/32) is always even here (24 or 96).
// mode 0: out_bf16 = C + bias
// mode 1: out_bf16 = gelu(C + bias)        (exact, erf)
// mode 2: out_f32  = C + bias + residual
// ---------------------------------------------------------------------------
#define TM 128
#define TN 128
#define TK 32
#define ASTRIDE (TM * (TK + 8))   // elements per A buffer

__global__ __launch_bounds__(256) void gemm_bf16_kernel(
    const __bf16* __restrict__ A, const __bf16* __restrict__ Bp,
    const float* __restrict__ bias, const float* __restrict__ res,
    __bf16* __restrict__ outb, float* __restrict__ outf,
    int M, int N, int K, int mode) {
  __shared__ __attribute__((aligned(16))) __bf16 As[2][TM][TK + 8];

  int tid  = threadIdx.x;
  int m0   = blockIdx.y * TM;
  int n0   = blockIdx.x * TN;
  int wave = tid >> 5;
  int wm   = wave & 3;     // 4 wave rows  * 32
  int wn   = wave >> 2;    // 2 wave cols  * 64
  int L    = tid & 31, lm = L & 15, g = L >> 4;
  int KB   = K >> 5;
  int nbw  = (n0 >> 4) + wn * 4;   // first 16-col block owned by this wave

  v8f acc[2][4];
#pragma unroll
  for (int i = 0; i < 2; ++i)
#pragma unroll
    for (int j = 0; j < 4; ++j)
#pragma unroll
      for (int e = 0; e < 8; ++e) acc[i][j][e] = 0.f;

  // A staging: 512 16B chunks, 2 per thread, pointer-stepped by TK.
  int ar = tid >> 2, ac = (tid & 3) * 8;
  gcbf_t ag0 = (gcbf_t)A + (size_t)(m0 + ar) * K + ac;
  gcbf_t ag1 = (gcbf_t)A + (size_t)(m0 + ar + 64) * K + ac;
  const __bf16* apre0 = A + (size_t)(m0 + ar) * K + ac;        // prefetch mirror
  const __bf16* apre1 = A + (size_t)(m0 + ar + 64) * K + ac;
  lbf_t  asb = (lbf_t)&As[0][0][0];
  int    aso0 = ar * (TK + 8) + ac;
  int    aso1 = (ar + 64) * (TK + 8) + ac;

  auto stageA = [&](int buf) {
    v4u d0 = *(gcu4_t)ag0;                     // global_load_b128
    v4u d1 = *(gcu4_t)ag1;
    int bo = buf * ASTRIDE;
    *(lu4_t)(asb + bo + aso0) = d0;            // ds_store_b128
    *(lu4_t)(asb + bo + aso1) = d1;
    ag0 += TK;
    ag1 += TK;
  };

  auto loadAfrag = [&](v16bf (&af)[2], int buf) {
#pragma unroll
    for (int i = 0; i < 2; ++i)
      af[i] = frag_row((lcbf_t)asb + buf * ASTRIDE +
                           (wm * 32 + i * 16) * (TK + 8),
                       TK + 8);
  };

  // B fragment pointers: one per owned 16-col block; step 512 elems / K-step.
  gcbf_t bptr[4];
#pragma unroll
  for (int j = 0; j < 4; ++j)
    bptr[j] = (gcbf_t)Bp + (size_t)(nbw + j) * KB * 512 + (size_t)L * 16;

  // Preload B fragments for kb = 0.
  v16bf b0[4], b1[4];
#pragma unroll
  for (int j = 0; j < 4; ++j) { b0[j] = *(gcv16_t)bptr[j]; bptr[j] += 512; }

  stageA(0);

  for (int kb = 0; kb < KB; kb += 2) {
    // ---- even sub-step: LDS buf 0, B regs b0; loads b1 for kb+1 ----
    __syncthreads();
    stageA(1);                                 // kb+1 < KB (KB even)
    if (kb + 2 < KB) {
      __builtin_prefetch(apre0 + (kb + 2) * TK, 0, 1);
      __builtin_prefetch(apre1 + (kb + 2) * TK, 0, 1);
    }
    v16bf af[2];
    loadAfrag(af, 0);
#pragma unroll
    for (int j = 0; j < 4; ++j) { b1[j] = *(gcv16_t)bptr[j]; bptr[j] += 512; }
#pragma unroll
    for (int j = 0; j < 4; ++j)
#pragma unroll
      for (int i = 0; i < 2; ++i)
        acc[i][j] = wmma_bf16(af[i], b0[j], acc[i][j]);

    // ---- odd sub-step: LDS buf 1, B regs b1; loads b0 for kb+2 ----
    __syncthreads();
    if (kb + 2 < KB) stageA(0);
    loadAfrag(af, 1);
    if (kb + 2 < KB) {
#pragma unroll
      for (int j = 0; j < 4; ++j) { b0[j] = *(gcv16_t)bptr[j]; bptr[j] += 512; }
    }
#pragma unroll
    for (int j = 0; j < 4; ++j)
#pragma unroll
      for (int i = 0; i < 2; ++i)
        acc[i][j] = wmma_bf16(af[i], b1[j], acc[i][j]);
  }

  // Epilogue. C layout: VGPR e -> M = e + 8*(lane/16); N = lane%16.
  gbf_t  outbg = (gbf_t)outb;
  gf_t   outfg = (gf_t)outf;
  gcf_t  resg  = (gcf_t)res;
  gcf_t  biasg = (gcf_t)bias;
#pragma unroll
  for (int i = 0; i < 2; ++i) {
#pragma unroll
    for (int j = 0; j < 4; ++j) {
      int col = n0 + wn * 64 + j * 16 + lm;
      float bv = bias ? biasg[col] : 0.f;
#pragma unroll
      for (int e = 0; e < 8; ++e) {
        int row = m0 + wm * 32 + i * 16 + g * 8 + e;
        size_t o = (size_t)row * N + col;
        float v = acc[i][j][e] + bv;
        if (mode == 0) {
          outbg[o] = (__bf16)v;
        } else if (mode == 1) {
          float t = 0.5f * v * (1.0f + erff(v * 0.70710678118f));
          outbg[o] = (__bf16)t;
        } else {
          outfg[o] = v + resg[o];
        }
      }
    }
  }
}

// ---------------------------------------------------------------------------
// Flash attention: block = 4 waves (128 thr) per (b, h, 64-query) tile.
// S = Q@K^T (row-major LDS, frags of two 16B runs), online softmax, O += P@V with
// P and V staged in LDS *already in fragment layout* (32B contiguous reads).
// ---------------------------------------------------------------------------
__global__ __launch_bounds__(128) void attention_kernel(
    const __bf16* __restrict__ qkv, const unsigned char* __restrict__ mask,
    __bf16* __restrict__ outb) {
  __shared__ __attribute__((aligned(16))) __bf16 Qs[64][72];
  __shared__ __attribute__((aligned(16))) __bf16 Ks[64][72];
  __shared__ __attribute__((aligned(16))) __bf16 Vf[2][4][32][24]; // frag-packed
  __shared__ __attribute__((aligned(16))) __bf16 Ps[4][2][32][24]; // frag-packed

  gcbf_t qkvg  = (gcbf_t)qkv;
  gcu8_t maskg = (gcu8_t)mask;
  int tid  = threadIdx.x;
  int qt   = blockIdx.x;                 // 0..31 query tiles
  int bh   = blockIdx.y;                 // 0..23
  int bi   = bh / NHEAD, h = bh % NHEAD;
  int tok0 = bi * SEQ + qt * 64;
  size_t hq = (size_t)h * DHEAD;
  const size_t kstep = (size_t)64 * QKVW;     // ptr step per key-block

  // Load Q tile (64x64 bf16) row-major, b128-wide.
  for (int c = tid; c < 512; c += 128) {
    int r = c >> 3, c8 = (c & 7) * 8;
    *(lu4_t)&Qs[r][c8] =
        *(gcu4_t)(qkvg + (size_t)(tok0 + r) * QKVW + hq + c8);
  }
  __syncthreads();

  int wave = tid >> 5;
  int L = tid & 31, lm = L & 15, g = L >> 4;

  v16bf aq[2];
#pragma unroll
  for (int ds = 0; ds < 2; ++ds)
    aq[ds] = frag_row((lcbf_t)&Qs[wave * 16][ds * 32], 72);

  // K staging: 4 b128 chunks per thread, pointer-stepped.
  gcbf_t kg[4];
  lu4_t  kd[4];
#pragma unroll
  for (int it = 0; it < 4; ++it) {
    int c = tid + it * 128;
    int r = c >> 3, c8 = (c & 7) * 8;
    kg[it] = qkvg + (size_t)(bi * SEQ + r) * QKVW + DIMX + hq + c8;
    kd[it] = (lu4_t)&Ks[r][c8];
  }
  // V staging (transpose into B-fragment layout): 4 chunks per thread.
  gcbf_t vg[4];
  lu4_t  vd[4];
#pragma unroll
  for (int it = 0; it < 4; ++it) {
    int c = tid + it * 128;
    int s  = c >> 8;
    int dj = (c >> 6) & 3;
    int Lv = (c >> 1) & 31;
    int hf = c & 1;
    int key = s * 32 + hf * 16 + (Lv >> 4) * 8;
    int d   = dj * 16 + (Lv & 15);
    vg[it] = qkvg + (size_t)(bi * SEQ + key) * QKVW + 2 * DIMX + hq + d;
    vd[it] = (lu4_t)&Vf[s][dj][Lv][hf * 8];
  }

  float mrow[8], lrow[8];
  v8f o[4];
#pragma unroll
  for (int i = 0; i < 8; ++i) { mrow[i] = -3.0e38f; lrow[i] = 0.f; }
#pragma unroll
  for (int j = 0; j < 4; ++j)
#pragma unroll
    for (int e = 0; e < 8; ++e) o[j][e] = 0.f;

  const float scale = 0.125f;  // DHEAD^-0.5

  for (int kb = 0; kb < SEQ / 64; ++kb) {
    __syncthreads();
#pragma unroll
    for (int it = 0; it < 4; ++it) {
      *kd[it] = *(gcu4_t)kg[it];
      kg[it] += kstep;
    }
#pragma unroll
    for (int it = 0; it < 4; ++it) {
      union { __bf16 h[8]; v4u v; } u;
#pragma unroll
      for (int i = 0; i < 8; ++i) u.h[i] = vg[it][(size_t)i * QKVW];
      *vd[it] = u.v;
      vg[it] += kstep;
    }
    __syncthreads();

    // S = Q @ K^T : K^T fragment = A-style read of row-major K tile.
    v8f s[4];
#pragma unroll
    for (int j = 0; j < 4; ++j) {
#pragma unroll
      for (int e = 0; e < 8; ++e) s[j][e] = 0.f;
#pragma unroll
      for (int ds = 0; ds < 2; ++ds) {
        v16bf bk = frag_row((lcbf_t)&Ks[j * 16][ds * 32], 72);
        s[j] = wmma_bf16(aq[ds], bk, s[j]);
      }
      int kidx = kb * 64 + j * 16 + lm;
      bool mk = maskg[bi * SEQ + kidx] != 0;
#pragma unroll
      for (int e = 0; e < 8; ++e)
        s[j][e] = mk ? s[j][e] * scale : -3.0e38f;
    }

    // Online softmax: row stats live across 16-lane halves (wave32).
    float alpha[8];
#pragma unroll
    for (int i = 0; i < 8; ++i) {
      float v = fmaxf(fmaxf(s[0][i], s[1][i]), fmaxf(s[2][i], s[3][i]));
#pragma unroll
      for (int off = 1; off < 16; off <<= 1)
        v = fmaxf(v, __shfl_xor(v, off, 32));
      float nm = fmaxf(mrow[i], v);
      alpha[i] = __expf(mrow[i] - nm);
      mrow[i] = nm;
    }
    float ls[8];
#pragma unroll
    for (int i = 0; i < 8; ++i) ls[i] = 0.f;
#pragma unroll
    for (int j = 0; j < 4; ++j)
#pragma unroll
      for (int i = 0; i < 8; ++i) {
        float p = __expf(s[j][i] - mrow[i]);
        s[j][i] = p;
        ls[i] += p;
      }
#pragma unroll
    for (int i = 0; i < 8; ++i) {
      float v = ls[i];
#pragma unroll
      for (int off = 1; off < 16; off <<= 1) v += __shfl_xor(v, off, 32);
      lrow[i] = lrow[i] * alpha[i] + v;
    }
#pragma unroll
    for (int j = 0; j < 4; ++j)
#pragma unroll
      for (int i = 0; i < 8; ++i) o[j][i] *= alpha[i];

    // P (C-frag) scattered into A-fragment layout in LDS.
#pragma unroll
    for (int j = 0; j < 4; ++j) {
      int k  = j * 16 + lm;
      int sb = k >> 5, kp = k & 31;
      int Lp = ((kp >> 3) & 1) << 4;
      int ep = ((kp >> 4) << 3) + (kp & 7);
      lbf_t pbase = (lbf_t)&Ps[wave][sb][Lp + 8 * g][ep];
#pragma unroll
      for (int i = 0; i < 8; ++i)
        pbase[i * 24] = (__bf16)s[j][i];
    }
    __syncthreads();

    // O += P @ V  (both operands are contiguous 32B fragment loads)
#pragma unroll
    for (int dj = 0; dj < 4; ++dj)
#pragma unroll
      for (int ks = 0; ks < 2; ++ks) {
        v16bf ap = *(lcv16_t)&Ps[wave][ks][L][0];
        v16bf bv = *(lcv16_t)&Vf[ks][dj][L][0];
        o[dj] = wmma_bf16(ap, bv, o[dj]);
      }
  }

  // Normalize and write bf16 output [token][768] at head offset.
  gbf_t outg = (gbf_t)outb;
#pragma unroll
  for (int dj = 0; dj < 4; ++dj)
#pragma unroll
    for (int i = 0; i < 8; ++i) {
      int row = tok0 + wave * 16 + 8 * g + i;
      int col = h * DHEAD + dj * 16 + lm;
      float v = o[dj][i] / lrow[i];
      outg[(size_t)row * DIMX + col] = (__bf16)v;
    }
}

// ---------------------------------------------------------------------------
// Host orchestration
// ---------------------------------------------------------------------------
extern "C" void kernel_launch(void* const* d_in, const int* in_sizes, int n_in,
                              void* d_out, int out_size, void* d_ws,
                              size_t ws_size, hipStream_t stream) {
  (void)in_sizes; (void)n_in; (void)out_size; (void)ws_size;

  const float*         x_in  = (const float*)d_in[0];
  const unsigned char* mask  = (const unsigned char*)d_in[1];   // bool array
  const float*         ln1_g = (const float*)d_in[2];
  const float*         ln1_b = (const float*)d_in[3];
  const float*         w_qkv = (const float*)d_in[4];
  const float*         w_out = (const float*)d_in[5];
  const float*         b_out = (const float*)d_in[6];
  const float*         ln2_g = (const float*)d_in[7];
  const float*         ln2_b = (const float*)d_in[8];
  const float*         w_ff1 = (const float*)d_in[9];
  const float*         b_ff1 = (const float*)d_in[10];
  const float*         w_ff2 = (const float*)d_in[11];
  const float*         b_ff2 = (const float*)d_in[12];
  const float*         lnf_g = (const float*)d_in[13];
  const float*         lnf_b = (const float*)d_in[14];

  char* ws = (char*)d_ws;
  size_t off = 0;
  float*  x      = (float*)(ws + off);  off += (size_t)NTOK * DIMX * sizeof(float);
  __bf16* lnbuf  = (__bf16*)(ws + off); off += (size_t)NTOK * DIMX * sizeof(__bf16);
  __bf16* ffbuf  = (__bf16*)(ws + off); off += (size_t)NTOK * FFW  * sizeof(__bf16);
  __bf16* qkvbf  = (__bf16*)(ws + off); off += (size_t)NTOK * QKVW * sizeof(__bf16);
  __bf16* attnbf = (__bf16*)(ws + off); off += (size_t)NTOK * DIMX * sizeof(__bf16);
  __bf16* wbuf   = (__bf16*)(ws + off); off += (size_t)DIMX * FFW  * sizeof(__bf16);

  (void)hipMemcpyAsync(x, x_in, (size_t)NTOK * DIMX * sizeof(float),
                       hipMemcpyDeviceToDevice, stream);

  for (int l = 0; l < 6; ++l) {
    // h = LN1(x) -> bf16
    ln_kernel<<<NTOK, 256, 0, stream>>>(x, ln1_g + l * DIMX, ln1_b + l * DIMX,
                                        lnbuf, nullptr, 0);
    // qkv = h @ w_qkv[l]  (no bias) -> bf16
    repack_w_kernel<<<dim3(QKVW / 16, DIMX / 32), 64, 0, stream>>>(
        w_qkv + (size_t)l * DIMX * QKVW, wbuf, DIMX, QKVW);
    gemm_bf16_kernel<<<dim3(QKVW / TN, NTOK / TM), 256, 0, stream>>>(
        lnbuf, wbuf, nullptr, nullptr, qkvbf, nullptr, NTOK, QKVW, DIMX, 0);
    // attention -> bf16 [token, 768]
    attention_kernel<<<dim3(SEQ / 64, 2 * NHEAD), 128, 0, stream>>>(
        qkvbf, mask, attnbf);
    // x = attn @ w_out[l] + b_out[l] + x
    repack_w_kernel<<<dim3(DIMX / 16, DIMX / 32), 64, 0, stream>>>(
        w_out + (size_t)l * DIMX * DIMX, wbuf, DIMX, DIMX);
    gemm_bf16_kernel<<<dim3(DIMX / TN, NTOK / TM), 256, 0, stream>>>(
        attnbf, wbuf, b_out + l * DIMX, x, nullptr, x, NTOK, DIMX, DIMX, 2);
    // h = LN2(x) -> bf16
    ln_kernel<<<NTOK, 256, 0, stream>>>(x, ln2_g + l * DIMX, ln2_b + l * DIMX,
                                        lnbuf, nullptr, 0);
    // h = gelu(h @ w_ff1[l] + b_ff1[l]) -> bf16
    repack_w_kernel<<<dim3(FFW / 16, DIMX / 32), 64, 0, stream>>>(
        w_ff1 + (size_t)l * DIMX * FFW, wbuf, DIMX, FFW);
    gemm_bf16_kernel<<<dim3(FFW / TN, NTOK / TM), 256, 0, stream>>>(
        lnbuf, wbuf, b_ff1 + l * FFW, nullptr, ffbuf, nullptr,
        NTOK, FFW, DIMX, 1);
    // x = h @ w_ff2[l] + b_ff2[l] + x
    repack_w_kernel<<<dim3(DIMX / 16, FFW / 32), 64, 0, stream>>>(
        w_ff2 + (size_t)l * FFW * DIMX, wbuf, FFW, DIMX);
    gemm_bf16_kernel<<<dim3(DIMX / TN, NTOK / TM), 256, 0, stream>>>(
        ffbuf, wbuf, b_ff2 + l * DIMX, x, nullptr, x, NTOK, DIMX, FFW, 2);
  }

  // final layer norm -> fp32 output
  ln_kernel<<<NTOK, 256, 0, stream>>>(x, lnf_g, lnf_b, nullptr,
                                      (float*)d_out, 1);
}
